// CMFuser_60043642798867
// MI455X (gfx1250) — matrix operand order repaired and checked by reading, not compiled, not run on hardware
//
#include <hip/hip_runtime.h>
#include <hip/hip_bf16.h>
#include <math.h>

// ---------- types ----------
typedef __bf16        bf16x16 __attribute__((ext_vector_type(16)));
typedef float         f32x8   __attribute__((ext_vector_type(8)));
typedef unsigned int  u32x4   __attribute__((ext_vector_type(4)));

union FragAB { u32x4 u[2]; bf16x16 v; };

__device__ __forceinline__ unsigned short f2bf(float x) {
  unsigned int u = __builtin_bit_cast(unsigned int, x);
  u += 0x7FFFu + ((u >> 16) & 1u);           // round-to-nearest-even
  return (unsigned short)(u >> 16);
}
__device__ __forceinline__ float bf2f(unsigned short h) {
  unsigned int u = ((unsigned int)h) << 16;
  return __builtin_bit_cast(float, u);
}

__device__ __forceinline__ float block_reduce_256(float v, float* sbuf) {
  int tid = threadIdx.x;
  sbuf[tid] = v;
  __syncthreads();
#pragma unroll
  for (int s = 128; s > 0; s >>= 1) {
    if (tid < s) sbuf[tid] += sbuf[tid + s];
    __syncthreads();
  }
  float r = sbuf[0];
  __syncthreads();
  return r;
}

// ---------- f32 -> bf16 convert (weights) ----------
__global__ void cvt_bf16_kernel(const float* __restrict__ in,
                                unsigned short* __restrict__ out, int n) {
  int i = blockIdx.x * 256 + threadIdx.x;
  if (i < n) out[i] = f2bf(in[i]);
}

// ---------- setup: BN fold + |gamma| top-K exchange coefficients ----------
// coef layout (8 x 768 f32): s_r, t_r, s_d, t_d, a_r, b_r, a_d, b_d
__global__ __launch_bounds__(256) void setup_kernel(
    const float* __restrict__ brw, const float* __restrict__ brb,
    const float* __restrict__ brm, const float* __restrict__ brv,
    const float* __restrict__ bdw, const float* __restrict__ bdb,
    const float* __restrict__ bdm, const float* __restrict__ bdv,
    const float* __restrict__ alpha, float* __restrict__ coef) {
  __shared__ float awr[768], awd[768];
  const int tid = threadIdx.x;
  for (int j = 0; j < 3; ++j) {
    int c = tid + j * 256;
    awr[c] = fabsf(brw[c]);
    awd[c] = fabsf(bdw[c]);
  }
  __syncthreads();
  for (int j = 0; j < 3; ++j) {
    int c = tid + j * 256;
    float sr = brw[c] * rsqrtf(brv[c] + 1e-5f);
    float sd = bdw[c] * rsqrtf(bdv[c] + 1e-5f);
    coef[c]          = sr;
    coef[768 + c]    = brb[c] - brm[c] * sr;
    coef[1536 + c]   = sd;
    coef[2304 + c]   = bdb[c] - bdm[c] * sd;
    // rank of |w_c| ascending, ties broken by lower index (top_k semantics)
    int rr = 0, rd = 0;
    float vr = awr[c], vd = awd[c];
    for (int jj = 0; jj < 768; ++jj) {
      rr += (awr[jj] < vr || (awr[jj] == vr && jj < c)) ? 1 : 0;
      rd += (awd[jj] < vd || (awd[jj] == vd && jj < c)) ? 1 : 0;
    }
    float al = alpha[c];
    bool mr = rr < 153, md = rd < 153;
    coef[3072 + c] = mr ? al : 1.f;
    coef[3840 + c] = mr ? (1.f - al) : 0.f;
    coef[4608 + c] = md ? al : 1.f;
    coef[5376 + c] = md ? (1.f - al) : 0.f;
  }
}

// ---------- fusion + LN1: x f32 (residual) and h bf16 ----------
__global__ __launch_bounds__(256) void fuse_ln1_kernel(
    const float* __restrict__ rgb, const float* __restrict__ depth,
    const float* __restrict__ coef,
    const float* __restrict__ w1, const float* __restrict__ b1,
    float* __restrict__ X, unsigned short* __restrict__ Hh) {
  __shared__ float sbuf[256];
  const int n = blockIdx.x, tid = threadIdx.x;
  float er[3], ed[3];
  float s0 = 0.f, q0 = 0.f, s1 = 0.f, q1 = 0.f;
  for (int j = 0; j < 3; ++j) {
    int c = tid + j * 256;
    float rn = rgb[(size_t)n * 768 + c] * coef[c] + coef[768 + c];
    float dn = depth[(size_t)n * 768 + c] * coef[1536 + c] + coef[2304 + c];
    float a = coef[3072 + c] * rn + coef[3840 + c] * dn;   // ex_rgb
    float b = coef[4608 + c] * dn + coef[5376 + c] * rn;   // ex_depth
    er[j] = a; ed[j] = b;
    s0 += a; q0 += a * a; s1 += b; q1 += b * b;
  }
  s0 = block_reduce_256(s0, sbuf);
  q0 = block_reduce_256(q0, sbuf);
  s1 = block_reduce_256(s1, sbuf);
  q1 = block_reduce_256(q1, sbuf);
  const float inv = 1.f / 768.f;
  float m0 = s0 * inv, m1 = s1 * inv;
  float r0 = rsqrtf(q0 * inv - m0 * m0 + 1e-5f);
  float r1 = rsqrtf(q1 * inv - m1 * m1 + 1e-5f);
  const size_t row0 = (size_t)(2 * n) * 768, row1 = row0 + 768;
  for (int j = 0; j < 3; ++j) {
    int c = tid + j * 256;
    X[row0 + c] = er[j];
    X[row1 + c] = ed[j];
    Hh[row0 + c] = f2bf((er[j] - m0) * r0 * w1[c] + b1[c]);
    Hh[row1 + c] = f2bf((ed[j] - m1) * r1 * w1[c] + b1[c]);
  }
}

// ---------- generic LN over rows -> bf16 ----------
__global__ __launch_bounds__(256) void ln_rows_kernel(
    const float* __restrict__ X, const float* __restrict__ w,
    const float* __restrict__ b, unsigned short* __restrict__ Hout) {
  __shared__ float sbuf[256];
  const size_t row = (size_t)blockIdx.x * 768;
  const int tid = threadIdx.x;
  float x[3];
  float s = 0.f, q = 0.f;
  for (int j = 0; j < 3; ++j) {
    x[j] = X[row + tid + j * 256];
    s += x[j]; q += x[j] * x[j];
  }
  s = block_reduce_256(s, sbuf);
  q = block_reduce_256(q, sbuf);
  const float inv = 1.f / 768.f;
  float m = s * inv;
  float r = rsqrtf(q * inv - m * m + 1e-5f);
  for (int j = 0; j < 3; ++j) {
    int c = tid + j * 256;
    Hout[row + c] = f2bf((x[j] - m) * r * w[c] + b[c]);
  }
}

// ---------- bf16 WMMA GEMM: out = act(A @ W^T + bias) (+ residual) ----------
// A: [M,K] bf16 row-major; W: [N,K] bf16 row-major.
// Block tile 64x256, K-step 32, double-buffered LDS (1 barrier / K-step).
// 8 waves in 2(M) x 4(N); each wave: 32x64 = 2x4 WMMA accumulators.
#define LDS_STRIDE 40   // halves per staged row (padded, conflict-free b128)
__global__ __launch_bounds__(256) void gemm_bf16_kernel(
    const unsigned short* __restrict__ A, const unsigned short* __restrict__ W,
    const float* __restrict__ bias, const float* __restrict__ residual,
    float* __restrict__ outF, unsigned short* __restrict__ outB,
    int M, int N, int K, int act) {
  __shared__ unsigned short sA[2][64 * LDS_STRIDE];
  __shared__ unsigned short sB[2][256 * LDS_STRIDE];
  const int tid = threadIdx.x;
  const int lane = tid & 31;
  const int wave = tid >> 5;
  const int wm = wave >> 2;                 // 0..1 (M sub-block of 32)
  const int wn = wave & 3;                  // 0..3 (N sub-block of 64)
  const int m0 = blockIdx.y * 64;
  const int n0 = blockIdx.x * 256;

  const int ar = tid >> 2, ac = (tid & 3) * 8;   // A stage: 64 rows x 32 halves
  const int br = tid;                             // B stage: 256 rows x 32 halves
  const int lr = lane & 15;
  const int hi = lane >> 4;
  const int aoff = hi ? 8 : 0;              // CDNA5 16-bit A 16x32 lane layout
  const int boff = hi ? 8 : 0;              // CDNA5 16-bit B 32x16 lane layout

  f32x8 acc[2][4] = {};

  const unsigned short* Arow = A + (size_t)(m0 + ar) * K + ac;
  const unsigned short* Wrow = W + (size_t)(n0 + br) * K;

  // ---- prologue: stage tile 0 into buffer 0 ----
  u32x4 av = *(const u32x4*)(Arow);
  u32x4 bvv[4];
#pragma unroll
  for (int q = 0; q < 4; ++q) bvv[q] = *(const u32x4*)(Wrow + q * 8);
  *(u32x4*)(sA[0] + ar * LDS_STRIDE + ac) = av;
#pragma unroll
  for (int q = 0; q < 4; ++q) *(u32x4*)(sB[0] + br * LDS_STRIDE + q * 8) = bvv[q];

  int buf = 0;
  for (int k0 = 0; k0 < K; k0 += 32) {
    const bool more = (k0 + 32) < K;
    if (more) {                              // prefetch tile i+1 (global)
      av = *(const u32x4*)(Arow + k0 + 32);
#pragma unroll
      for (int q = 0; q < 4; ++q) bvv[q] = *(const u32x4*)(Wrow + k0 + 32 + q * 8);
    }
    __syncthreads();                         // stage i visible; buf^1 free
    // fragment loads from buffer `buf` (tile i)
    FragAB afr[2], bfr[4];
#pragma unroll
    for (int mt = 0; mt < 2; ++mt) {
      const unsigned short* p = sA[buf] + (wm * 32 + mt * 16 + lr) * LDS_STRIDE + aoff;
      afr[mt].u[0] = *(const u32x4*)(p);        // K 0..7   (or 8..15)
      afr[mt].u[1] = *(const u32x4*)(p + 16);   // K 16..23 (or 24..31)
    }
#pragma unroll
    for (int nt = 0; nt < 4; ++nt) {
      const unsigned short* p = sB[buf] + (wn * 64 + nt * 16 + lr) * LDS_STRIDE + boff;
      bfr[nt].u[0] = *(const u32x4*)(p);        // K 0..7   (or 8..15)
      bfr[nt].u[1] = *(const u32x4*)(p + 16);   // K 16..23 (or 24..31)
    }
    if (more) {                              // stage tile i+1 into buf^1
      *(u32x4*)(sA[buf ^ 1] + ar * LDS_STRIDE + ac) = av;
#pragma unroll
      for (int q = 0; q < 4; ++q)
        *(u32x4*)(sB[buf ^ 1] + br * LDS_STRIDE + q * 8) = bvv[q];
    }
#pragma unroll
    for (int mt = 0; mt < 2; ++mt)
#pragma unroll
      for (int nt = 0; nt < 4; ++nt)
        acc[mt][nt] = __builtin_amdgcn_wmma_f32_16x16x32_bf16(
            false, afr[mt].v, false, bfr[nt].v, (short)0, acc[mt][nt],
            false, false);
    buf ^= 1;
  }

#pragma unroll
  for (int mt = 0; mt < 2; ++mt) {
#pragma unroll
    for (int nt = 0; nt < 4; ++nt) {
      const int col = n0 + wn * 64 + nt * 16 + lr;
      const float bv = bias ? bias[col] : 0.f;
#pragma unroll
      for (int r2 = 0; r2 < 8; ++r2) {
        const int row = m0 + wm * 32 + mt * 16 + hi * 8 + r2;
        float v = acc[mt][nt][r2] + bv;
        if (act == 1) v = 0.5f * v * (1.0f + erff(v * 0.70710678118654752f));
        if (residual) v += residual[(size_t)row * N + col];
        if (outF) outF[(size_t)row * N + col] = v;
        if (outB) outB[(size_t)row * N + col] = f2bf(v);
      }
    }
  }
}

// ---------- 2-token masked attention (diag = -1e9) ----------
// QKV: [2N, 2304] bf16, cols = {q,k,v} x {h} x {d}. O: [2N, 768] bf16.
__global__ __launch_bounds__(256) void attn_kernel(
    const unsigned short* __restrict__ QKV, unsigned short* __restrict__ O) {
  __shared__ float wgt[8][16];
  const int lane = threadIdx.x & 31;
  const int wv = threadIdx.x >> 5;
  const int n = blockIdx.x * 8 + wv;
  const size_t base = (size_t)n * 2 * 2304;
  if (lane < 16) {
    const int h = lane >> 2, m = (lane >> 1) & 1, kk = lane & 1;
    const u32x4* qp = (const u32x4*)(QKV + base + (size_t)m * 2304 + h * 192);
    const u32x4* kp = (const u32x4*)(QKV + base + (size_t)kk * 2304 + 768 + h * 192);
    float s = 0.f;
    for (int it = 0; it < 24; ++it) {
      u32x4 qv = qp[it], kv = kp[it];
#pragma unroll
      for (int e = 0; e < 4; ++e) {
        s += bf2f((unsigned short)(qv[e] & 0xffffu)) * bf2f((unsigned short)(kv[e] & 0xffffu));
        s += bf2f((unsigned short)(qv[e] >> 16)) * bf2f((unsigned short)(kv[e] >> 16));
      }
    }
    s *= 0.07216878364870323f;               // 192^-0.5
    if (m == kk) s += -1e9f;
    float so = __shfl_xor(s, 1, 32);
    float mx = fmaxf(s, so);
    float e0 = expf(s - mx), e1 = expf(so - mx);
    wgt[wv][lane] = e0 / (e0 + e1);
  }
  __syncthreads();
  for (int m = 0; m < 2; ++m) {
    for (int d = lane; d < 768; d += 32) {
      const int h = d / 192;
      const float w0 = wgt[wv][h * 4 + m * 2 + 0];
      const float w1 = wgt[wv][h * 4 + m * 2 + 1];
      const float v0 = bf2f(QKV[base + 1536 + d]);
      const float v1 = bf2f(QKV[base + 2304 + 1536 + d]);
      O[(size_t)(2 * n + m) * 768 + d] = f2bf(w0 * v0 + w1 * v1);
    }
  }
}

// ---------- final LN + mean over modality tokens ----------
__global__ __launch_bounds__(256) void final_kernel(
    const float* __restrict__ X, const float* __restrict__ w,
    const float* __restrict__ b, float* __restrict__ out) {
  __shared__ float sbuf[256];
  const int n = blockIdx.x, tid = threadIdx.x;
  const size_t r0 = (size_t)(2 * n) * 768, r1 = r0 + 768;
  float x0[3], x1[3];
  float s0 = 0.f, q0 = 0.f, s1 = 0.f, q1 = 0.f;
  for (int j = 0; j < 3; ++j) {
    int c = tid + j * 256;
    x0[j] = X[r0 + c]; x1[j] = X[r1 + c];
    s0 += x0[j]; q0 += x0[j] * x0[j];
    s1 += x1[j]; q1 += x1[j] * x1[j];
  }
  s0 = block_reduce_256(s0, sbuf);
  q0 = block_reduce_256(q0, sbuf);
  s1 = block_reduce_256(s1, sbuf);
  q1 = block_reduce_256(q1, sbuf);
  const float inv = 1.f / 768.f;
  float m0 = s0 * inv, m1 = s1 * inv;
  float i0 = rsqrtf(q0 * inv - m0 * m0 + 1e-5f);
  float i1 = rsqrtf(q1 * inv - m1 * m1 + 1e-5f);
  for (int j = 0; j < 3; ++j) {
    int c = tid + j * 256;
    float y0 = (x0[j] - m0) * i0 * w[c] + b[c];
    float y1 = (x1[j] - m1) * i1 * w[c] + b[c];
    out[(size_t)n * 768 + c] = 0.5f * (y0 + y1);
  }
}

extern "C" void kernel_launch(void* const* d_in, const int* in_sizes, int n_in,
                              void* d_out, int out_size, void* d_ws, size_t ws_size,
                              hipStream_t stream) {
  (void)in_sizes; (void)n_in; (void)out_size; (void)ws_size;
  const float* rgb     = (const float*)d_in[0];
  const float* depth   = (const float*)d_in[1];
  const float* brw     = (const float*)d_in[2];
  const float* brb     = (const float*)d_in[3];
  const float* brm     = (const float*)d_in[4];
  const float* brv     = (const float*)d_in[5];
  const float* bdw     = (const float*)d_in[6];
  const float* bdb     = (const float*)d_in[7];
  const float* bdm     = (const float*)d_in[8];
  const float* bdv     = (const float*)d_in[9];
  const float* alpha   = (const float*)d_in[10];
  const float* norm1_w = (const float*)d_in[11];
  const float* norm1_b = (const float*)d_in[12];
  const float* qkv_w   = (const float*)d_in[13];
  const float* proj_w  = (const float*)d_in[14];
  const float* proj_b  = (const float*)d_in[15];
  const float* norm2_w = (const float*)d_in[16];
  const float* norm2_b = (const float*)d_in[17];
  const float* fc1_w   = (const float*)d_in[18];
  const float* fc1_b   = (const float*)d_in[19];
  const float* fc2_w   = (const float*)d_in[20];
  const float* fc2_b   = (const float*)d_in[21];
  const float* normf_w = (const float*)d_in[22];
  const float* normf_b = (const float*)d_in[23];

  const size_t N2 = 32768;                 // 2 * B * T token rows

  char* ws = (char*)d_ws;
  size_t off = 0;
  auto take = [&](size_t bytes) -> char* {
    char* p = ws + off;
    off += (bytes + 255) & ~(size_t)255;
    return p;
  };
  unsigned short* wQKV  = (unsigned short*)take((size_t)2304 * 768 * 2);
  unsigned short* wPROJ = (unsigned short*)take((size_t)768 * 768 * 2);
  unsigned short* wFC1  = (unsigned short*)take((size_t)3072 * 768 * 2);
  unsigned short* wFC2  = (unsigned short*)take((size_t)768 * 3072 * 2);
  float*          coef  = (float*)take((size_t)8 * 768 * 4);
  float*          bufX  = (float*)take(N2 * 768 * 4);           // residual stream
  unsigned short* bufH  = (unsigned short*)take(N2 * 768 * 2);  // LN out / attn out
  unsigned short* bufG  = (unsigned short*)take(N2 * 3072 * 2); // qkv out / gelu out

  // weight conversion (cheap; repeated per call for determinism)
  cvt_bf16_kernel<<<(2304 * 768 + 255) / 256, 256, 0, stream>>>(qkv_w, wQKV, 2304 * 768);
  cvt_bf16_kernel<<<(768 * 768 + 255) / 256, 256, 0, stream>>>(proj_w, wPROJ, 768 * 768);
  cvt_bf16_kernel<<<(3072 * 768 + 255) / 256, 256, 0, stream>>>(fc1_w, wFC1, 3072 * 768);
  cvt_bf16_kernel<<<(768 * 3072 + 255) / 256, 256, 0, stream>>>(fc2_w, wFC2, 768 * 3072);

  setup_kernel<<<1, 256, 0, stream>>>(brw, brb, brm, brv, bdw, bdb, bdm, bdv, alpha, coef);

  fuse_ln1_kernel<<<16384, 256, 0, stream>>>(rgb, depth, coef, norm1_w, norm1_b, bufX, bufH);

  // qkv: [32768,768] x [2304,768]^T -> bufG bf16
  gemm_bf16_kernel<<<dim3(2304 / 256, N2 / 64), 256, 0, stream>>>(
      bufH, wQKV, nullptr, nullptr, nullptr, bufG, (int)N2, 2304, 768, 0);

  attn_kernel<<<16384 / 8, 256, 0, stream>>>(bufG, bufH);

  // proj + residual: bufX += o @ proj^T + b
  gemm_bf16_kernel<<<dim3(768 / 256, N2 / 64), 256, 0, stream>>>(
      bufH, wPROJ, proj_b, bufX, bufX, nullptr, (int)N2, 768, 768, 0);

  ln_rows_kernel<<<(int)N2, 256, 0, stream>>>(bufX, norm2_w, norm2_b, bufH);

  // fc1 + exact GELU -> bufG bf16
  gemm_bf16_kernel<<<dim3(3072 / 256, N2 / 64), 256, 0, stream>>>(
      bufH, wFC1, fc1_b, nullptr, nullptr, bufG, (int)N2, 3072, 768, 1);

  // fc2 + residual: bufX += g @ fc2^T + b
  gemm_bf16_kernel<<<dim3(768 / 256, N2 / 64), 256, 0, stream>>>(
      bufG, wFC2, fc2_b, bufX, bufX, nullptr, (int)N2, 768, 3072, 0);

  final_kernel<<<16384, 256, 0, stream>>>(bufX, normf_w, normf_b, (float*)d_out);
}
